// LSTMWithHistory_36679020708035
// MI455X (gfx1250) — matrix operand order repaired and checked by reading, not compiled, not run on hardware
//
#include <hip/hip_runtime.h>
#include <hip/hip_bf16.h>
#include <math.h>

#define B_ 64
#define S_ 512
#define I_ 1024
#define H_ 1024
#define O_ 256
#define G4_ (4 * H_)

typedef __attribute__((ext_vector_type(16))) _Float16 v16h;
typedef __attribute__((ext_vector_type(8)))  _Float16 v8h;
typedef __attribute__((ext_vector_type(8)))  float    v8f;

__device__ __forceinline__ v16h cat8(v8h lo, v8h hi) {
    return __builtin_shufflevector(lo, hi, 0,1,2,3,4,5,6,7,8,9,10,11,12,13,14,15);
}
__device__ __forceinline__ float sigmoid_(float x) { return 1.0f / (1.0f + __expf(-x)); }

// -------- fp32 -> fp16 conversion (one-time) --------
__global__ void cvt_f16_kernel(const float* __restrict__ in, _Float16* __restrict__ out, long n) {
    long i  = (long)blockIdx.x * blockDim.x + threadIdx.x;
    long st = (long)gridDim.x * blockDim.x;
    for (; i < n; i += st) out[i] = (_Float16)in[i];
}

__global__ void bias_sum_kernel(const float* __restrict__ a, const float* __restrict__ b,
                                float* __restrict__ out, int n) {
    int i = blockIdx.x * blockDim.x + threadIdx.x;
    if (i < n) out[i] = a[i] + b[i];
}

__global__ void init_sync_kernel(int* p) { p[0] = 0; p[1] = 0; }

// -------- software grid barrier (64 blocks x 128 threads: always co-resident) --------
__device__ __forceinline__ void grid_sync(int* count, int* gen, int nblk) {
    __syncthreads();
    if (threadIdx.x == 0) {
        __threadfence();
        int g = __hip_atomic_load(gen, __ATOMIC_RELAXED, __HIP_MEMORY_SCOPE_AGENT);
        int arrived = __hip_atomic_fetch_add(count, 1, __ATOMIC_ACQ_REL, __HIP_MEMORY_SCOPE_AGENT);
        if (arrived == nblk - 1) {
            __hip_atomic_store(count, 0, __ATOMIC_RELAXED, __HIP_MEMORY_SCOPE_AGENT);
            __hip_atomic_fetch_add(gen, 1, __ATOMIC_RELEASE, __HIP_MEMORY_SCOPE_AGENT);
        } else {
            while (__hip_atomic_load(gen, __ATOMIC_ACQUIRE, __HIP_MEMORY_SCOPE_AGENT) == g) {
                __builtin_amdgcn_s_sleep(2);
            }
        }
        __threadfence();
    }
    __syncthreads();
}

// ============================================================================
// xg precompute: xg[s,b,:] = x[b,s,:] @ W_ih^T  for all 512 steps in parallel.
// grid: (4H/256, S*B/16) = (16, 2048) blocks, 128 thr = 4 waves (64 cols each).
// Output stored fragment-packed: [(s*4+rb)*256 + col_tile][lane][r] fp16, so the
// recurrence seeds each 16x16 accumulator with one b128 load per lane.
// ============================================================================
__global__ __launch_bounds__(128) void xg_precompute_kernel(
    const _Float16* __restrict__ xf,   // [B,S,I] f16
    const _Float16* __restrict__ Wih,  // [4H,I]  f16 row-major
    _Float16*       __restrict__ xgf)  // packed, S*B*4H elements
{
    const int tid  = threadIdx.x;
    const int wave = tid >> 5;
    const int lane = tid & 31;
    const int m    = lane & 15;
    const int half = lane >> 4;
    const int mt   = blockIdx.y;          // M-tile: s = mt>>2, batch-tile rb = mt&3
    const int s    = mt >> 2;
    const int b0   = (mt & 3) << 4;
    const int nb   = (blockIdx.x * 4 + wave) * 64;

    v8f a0 = {}, a1 = {}, a2 = {}, a3 = {};

    const _Float16* ar = xf + ((size_t)(b0 + m) * S_ + (size_t)s) * I_;
    const _Float16* w0 = Wih + (size_t)(nb +  0 + m) * I_ + half * 16;
    const _Float16* w1 = Wih + (size_t)(nb + 16 + m) * I_ + half * 16;
    const _Float16* w2 = Wih + (size_t)(nb + 32 + m) * I_ + half * 16;
    const _Float16* w3 = Wih + (size_t)(nb + 48 + m) * I_ + half * 16;
    #pragma unroll 2
    for (int k0 = 0; k0 < I_; k0 += 32) {
        v8h lo = *(const v8h*)(ar + k0 + half * 16);
        v8h hi = *(const v8h*)(ar + k0 + 8 + half * 16);
        v16h a  = cat8(lo, hi);
        v16h bv0 = *(const v16h*)(w0 + k0);
        v16h bv1 = *(const v16h*)(w1 + k0);
        v16h bv2 = *(const v16h*)(w2 + k0);
        v16h bv3 = *(const v16h*)(w3 + k0);
        a0 = __builtin_amdgcn_wmma_f32_16x16x32_f16(false, a, false, bv0, (short)0, a0, false, false);
        a1 = __builtin_amdgcn_wmma_f32_16x16x32_f16(false, a, false, bv1, (short)0, a1, false, false);
        a2 = __builtin_amdgcn_wmma_f32_16x16x32_f16(false, a, false, bv2, (short)0, a2, false, false);
        a3 = __builtin_amdgcn_wmma_f32_16x16x32_f16(false, a, false, bv3, (short)0, a3, false, false);
    }

    // pack-store each accumulator: 8 f16 contiguous per lane
    const size_t tg0 = (size_t)mt * 256 + (nb >> 4);
    v8f accs[4] = {a0, a1, a2, a3};
    #pragma unroll
    for (int t = 0; t < 4; ++t) {
        v8h o;
        #pragma unroll
        for (int r = 0; r < 8; ++r) o[r] = (_Float16)accs[t][r];
        *(v8h*)(xgf + ((tg0 + t) * 32 + lane) * 8) = o;
    }
}

// ============================================================================
// Persistent recurrence (xg path): serial path is only the h-GEMM (K=1024).
// grid: (H/64, B/16) = 64 blocks, 128 thr = 4 waves; wave w owns gate group w.
// ============================================================================
__global__ __launch_bounds__(128) void lstm_persistent_xg_kernel(
    const _Float16* __restrict__ xgf,   // packed xg
    const _Float16* __restrict__ Whh,   // [4H,H] f16 row-major
    const float*    __restrict__ bias,  // [4H]
    _Float16*       __restrict__ hb0,
    _Float16*       __restrict__ hb1,
    int*            __restrict__ sync_count,
    int*            __restrict__ sync_gen)
{
    __shared__ float gl[4][16][65];

    const int tid  = threadIdx.x;
    const int wave = tid >> 5;
    const int lane = tid & 31;
    const int m    = lane & 15;
    const int half = lane >> 4;
    const int b0   = blockIdx.y << 4;
    const int j0   = blockIdx.x << 6;
    const int nb   = wave * H_ + j0;
    const int nblk = gridDim.x * gridDim.y;
    const int rbk  = blockIdx.y;               // batch tile index
    const int tb   = wave * 64 + (j0 >> 4);    // global column-tile base

    const int col  = tid & 63;
    const int row0 = tid >> 6;
    float creg[8];
    #pragma unroll
    for (int k = 0; k < 8; ++k) creg[k] = 0.0f;
    const float bi = bias[0 * H_ + j0 + col];
    const float bf = bias[1 * H_ + j0 + col];
    const float bg = bias[2 * H_ + j0 + col];
    const float bo = bias[3 * H_ + j0 + col];

    const _Float16* wh0 = Whh + (size_t)(nb +  0 + m) * H_ + half * 16;
    const _Float16* wh1 = Whh + (size_t)(nb + 16 + m) * H_ + half * 16;
    const _Float16* wh2 = Whh + (size_t)(nb + 32 + m) * H_ + half * 16;
    const _Float16* wh3 = Whh + (size_t)(nb + 48 + m) * H_ + half * 16;

    for (int s = 0; s < S_; ++s) {
        const _Float16* hin  = (s & 1) ? hb1 : hb0;
        _Float16*       hout = (s & 1) ? hb0 : hb1;

        // ---- seed accumulators from packed xg: 4x b128 per lane ----
        const _Float16* xp = xgf + (((size_t)(s * 4 + rbk) * 256 + tb) * 32 + lane) * 8;
        v8h x0 = *(const v8h*)(xp + 0 * 256);
        v8h x1 = *(const v8h*)(xp + 1 * 256);
        v8h x2 = *(const v8h*)(xp + 2 * 256);
        v8h x3 = *(const v8h*)(xp + 3 * 256);
        v8f a0, a1, a2, a3;
        #pragma unroll
        for (int r = 0; r < 8; ++r) {
            a0[r] = (float)x0[r];
            a1[r] = (float)x1[r];
            a2[r] = (float)x2[r];
            a3[r] = (float)x3[r];
        }

        // ---- h contribution (K = H); skipped on step 0 where h == 0 ----
        if (s > 0) {
            const _Float16* ar = hin + (size_t)(b0 + m) * H_;
            #pragma unroll 2
            for (int k0 = 0; k0 < H_; k0 += 32) {
                v8h lo = *(const v8h*)(ar + k0 + half * 16);
                v8h hi = *(const v8h*)(ar + k0 + 8 + half * 16);
                v16h a  = cat8(lo, hi);
                v16h bv0 = *(const v16h*)(wh0 + k0);
                v16h bv1 = *(const v16h*)(wh1 + k0);
                v16h bv2 = *(const v16h*)(wh2 + k0);
                v16h bv3 = *(const v16h*)(wh3 + k0);
                a0 = __builtin_amdgcn_wmma_f32_16x16x32_f16(false, a, false, bv0, (short)0, a0, false, false);
                a1 = __builtin_amdgcn_wmma_f32_16x16x32_f16(false, a, false, bv1, (short)0, a1, false, false);
                a2 = __builtin_amdgcn_wmma_f32_16x16x32_f16(false, a, false, bv2, (short)0, a2, false, false);
                a3 = __builtin_amdgcn_wmma_f32_16x16x32_f16(false, a, false, bv3, (short)0, a3, false, false);
            }
        }

        // ---- spill gate tiles to LDS in C-layout ----
        #pragma unroll
        for (int r = 0; r < 8; ++r) {
            gl[wave][half * 8 + r][ 0 + m] = a0[r];
            gl[wave][half * 8 + r][16 + m] = a1[r];
            gl[wave][half * 8 + r][32 + m] = a2[r];
            gl[wave][half * 8 + r][48 + m] = a3[r];
        }
        __syncthreads();

        // ---- cell update: c register-resident across all steps ----
        #pragma unroll
        for (int k = 0; k < 8; ++k) {
            const int row = row0 + 2 * k;
            float gi = gl[0][row][col] + bi;
            float gf = gl[1][row][col] + bf;
            float gg = gl[2][row][col] + bg;
            float go = gl[3][row][col] + bo;
            float iv = sigmoid_(gi);
            float fv = sigmoid_(gf);
            float gv = tanhf(gg);
            float ov = sigmoid_(go);
            float cn = fv * creg[k] + iv * gv;
            creg[k] = cn;
            hout[(size_t)(b0 + row) * H_ + j0 + col] = (_Float16)(ov * tanhf(cn));
        }

        grid_sync(sync_count, sync_gen, nblk);
    }
}

// ============================================================================
// Fallback: fused persistent kernel (x-GEMM + h-GEMM per step), used when the
// workspace cannot hold the 256 MB packed xg buffer.
// ============================================================================
__global__ __launch_bounds__(128) void lstm_persistent_kernel(
    const _Float16* __restrict__ xf,
    const _Float16* __restrict__ Wih,
    const _Float16* __restrict__ Whh,
    const float*    __restrict__ bias,
    _Float16*       __restrict__ hb0,
    _Float16*       __restrict__ hb1,
    int*            __restrict__ sync_count,
    int*            __restrict__ sync_gen)
{
    __shared__ float gl[4][16][65];

    const int tid  = threadIdx.x;
    const int wave = tid >> 5;
    const int lane = tid & 31;
    const int m    = lane & 15;
    const int half = lane >> 4;
    const int b0   = blockIdx.y << 4;
    const int j0   = blockIdx.x << 6;
    const int nb   = wave * H_ + j0;
    const int nblk = gridDim.x * gridDim.y;

    const int col  = tid & 63;
    const int row0 = tid >> 6;
    float creg[8];
    #pragma unroll
    for (int k = 0; k < 8; ++k) creg[k] = 0.0f;
    const float bi = bias[0 * H_ + j0 + col];
    const float bf = bias[1 * H_ + j0 + col];
    const float bg = bias[2 * H_ + j0 + col];
    const float bo = bias[3 * H_ + j0 + col];

    const _Float16* xbase = xf + (size_t)(b0 + m) * S_ * I_;
    const _Float16* wi0 = Wih + (size_t)(nb +  0 + m) * I_ + half * 16;
    const _Float16* wi1 = Wih + (size_t)(nb + 16 + m) * I_ + half * 16;
    const _Float16* wi2 = Wih + (size_t)(nb + 32 + m) * I_ + half * 16;
    const _Float16* wi3 = Wih + (size_t)(nb + 48 + m) * I_ + half * 16;
    const _Float16* wh0 = Whh + (size_t)(nb +  0 + m) * H_ + half * 16;
    const _Float16* wh1 = Whh + (size_t)(nb + 16 + m) * H_ + half * 16;
    const _Float16* wh2 = Whh + (size_t)(nb + 32 + m) * H_ + half * 16;
    const _Float16* wh3 = Whh + (size_t)(nb + 48 + m) * H_ + half * 16;

    for (int s = 0; s < S_; ++s) {
        const _Float16* hin  = (s & 1) ? hb1 : hb0;
        _Float16*       hout = (s & 1) ? hb0 : hb1;

        v8f a0 = {}, a1 = {}, a2 = {}, a3 = {};
        {
            const _Float16* ar = xbase + (size_t)s * I_;
            #pragma unroll 2
            for (int k0 = 0; k0 < I_; k0 += 32) {
                v8h lo = *(const v8h*)(ar + k0 + half * 16);
                v8h hi = *(const v8h*)(ar + k0 + 8 + half * 16);
                v16h a  = cat8(lo, hi);
                v16h bv0 = *(const v16h*)(wi0 + k0);
                v16h bv1 = *(const v16h*)(wi1 + k0);
                v16h bv2 = *(const v16h*)(wi2 + k0);
                v16h bv3 = *(const v16h*)(wi3 + k0);
                a0 = __builtin_amdgcn_wmma_f32_16x16x32_f16(false, a, false, bv0, (short)0, a0, false, false);
                a1 = __builtin_amdgcn_wmma_f32_16x16x32_f16(false, a, false, bv1, (short)0, a1, false, false);
                a2 = __builtin_amdgcn_wmma_f32_16x16x32_f16(false, a, false, bv2, (short)0, a2, false, false);
                a3 = __builtin_amdgcn_wmma_f32_16x16x32_f16(false, a, false, bv3, (short)0, a3, false, false);
            }
        }
        if (s > 0) {
            const _Float16* ar = hin + (size_t)(b0 + m) * H_;
            #pragma unroll 2
            for (int k0 = 0; k0 < H_; k0 += 32) {
                v8h lo = *(const v8h*)(ar + k0 + half * 16);
                v8h hi = *(const v8h*)(ar + k0 + 8 + half * 16);
                v16h a  = cat8(lo, hi);
                v16h bv0 = *(const v16h*)(wh0 + k0);
                v16h bv1 = *(const v16h*)(wh1 + k0);
                v16h bv2 = *(const v16h*)(wh2 + k0);
                v16h bv3 = *(const v16h*)(wh3 + k0);
                a0 = __builtin_amdgcn_wmma_f32_16x16x32_f16(false, a, false, bv0, (short)0, a0, false, false);
                a1 = __builtin_amdgcn_wmma_f32_16x16x32_f16(false, a, false, bv1, (short)0, a1, false, false);
                a2 = __builtin_amdgcn_wmma_f32_16x16x32_f16(false, a, false, bv2, (short)0, a2, false, false);
                a3 = __builtin_amdgcn_wmma_f32_16x16x32_f16(false, a, false, bv3, (short)0, a3, false, false);
            }
        }

        #pragma unroll
        for (int r = 0; r < 8; ++r) {
            gl[wave][half * 8 + r][ 0 + m] = a0[r];
            gl[wave][half * 8 + r][16 + m] = a1[r];
            gl[wave][half * 8 + r][32 + m] = a2[r];
            gl[wave][half * 8 + r][48 + m] = a3[r];
        }
        __syncthreads();

        #pragma unroll
        for (int k = 0; k < 8; ++k) {
            const int row = row0 + 2 * k;
            float gi = gl[0][row][col] + bi;
            float gf = gl[1][row][col] + bf;
            float gg = gl[2][row][col] + bg;
            float go = gl[3][row][col] + bo;
            float iv = sigmoid_(gi);
            float fv = sigmoid_(gf);
            float gv = tanhf(gg);
            float ov = sigmoid_(go);
            float cn = fv * creg[k] + iv * gv;
            creg[k] = cn;
            hout[(size_t)(b0 + row) * H_ + j0 + col] = (_Float16)(ov * tanhf(cn));
        }

        grid_sync(sync_count, sync_gen, nblk);
    }
}

// -------- final FC: out = h_T @ W_fc^T + b_fc --------
__global__ __launch_bounds__(128) void fc_kernel(
    const _Float16* __restrict__ h,
    const _Float16* __restrict__ Wfc,
    const float*    __restrict__ bfc,
    float*          __restrict__ out)
{
    const int tid  = threadIdx.x;
    const int wave = tid >> 5;
    const int lane = tid & 31;
    const int m    = lane & 15;
    const int half = lane >> 4;
    const int b0   = blockIdx.x << 4;
    const int nb   = wave * 64;

    v8f a0 = {}, a1 = {}, a2 = {}, a3 = {};

    const _Float16* ar = h + (size_t)(b0 + m) * H_;
    const _Float16* w0 = Wfc + (size_t)(nb +  0 + m) * H_ + half * 16;
    const _Float16* w1 = Wfc + (size_t)(nb + 16 + m) * H_ + half * 16;
    const _Float16* w2 = Wfc + (size_t)(nb + 32 + m) * H_ + half * 16;
    const _Float16* w3 = Wfc + (size_t)(nb + 48 + m) * H_ + half * 16;
    #pragma unroll 2
    for (int k0 = 0; k0 < H_; k0 += 32) {
        v8h lo = *(const v8h*)(ar + k0 + half * 16);
        v8h hi = *(const v8h*)(ar + k0 + 8 + half * 16);
        v16h a  = cat8(lo, hi);
        v16h bv0 = *(const v16h*)(w0 + k0);
        v16h bv1 = *(const v16h*)(w1 + k0);
        v16h bv2 = *(const v16h*)(w2 + k0);
        v16h bv3 = *(const v16h*)(w3 + k0);
        a0 = __builtin_amdgcn_wmma_f32_16x16x32_f16(false, a, false, bv0, (short)0, a0, false, false);
        a1 = __builtin_amdgcn_wmma_f32_16x16x32_f16(false, a, false, bv1, (short)0, a1, false, false);
        a2 = __builtin_amdgcn_wmma_f32_16x16x32_f16(false, a, false, bv2, (short)0, a2, false, false);
        a3 = __builtin_amdgcn_wmma_f32_16x16x32_f16(false, a, false, bv3, (short)0, a3, false, false);
    }

    #pragma unroll
    for (int r = 0; r < 8; ++r) {
        const int row = b0 + half * 8 + r;
        out[(size_t)row * O_ + nb +  0 + m] = a0[r] + bfc[nb +  0 + m];
        out[(size_t)row * O_ + nb + 16 + m] = a1[r] + bfc[nb + 16 + m];
        out[(size_t)row * O_ + nb + 32 + m] = a2[r] + bfc[nb + 32 + m];
        out[(size_t)row * O_ + nb + 48 + m] = a3[r] + bfc[nb + 48 + m];
    }
}

extern "C" void kernel_launch(void* const* d_in, const int* in_sizes, int n_in,
                              void* d_out, int out_size, void* d_ws, size_t ws_size,
                              hipStream_t stream) {
    const float* x   = (const float*)d_in[0];
    const float* Wih = (const float*)d_in[1];
    const float* Whh = (const float*)d_in[2];
    const float* bih = (const float*)d_in[3];
    const float* bhh = (const float*)d_in[4];
    const float* Wfc = (const float*)d_in[5];
    const float* bfc = (const float*)d_in[6];
    float* out = (float*)d_out;

    char* ws = (char*)d_ws;
    size_t off = 0;
    auto alloc = [&](size_t bytes) -> char* {
        char* p = ws + off;
        off += (bytes + 255) & ~(size_t)255;
        return p;
    };
    _Float16* xf   = (_Float16*)alloc((size_t)B_ * S_ * I_ * 2);
    _Float16* wihf = (_Float16*)alloc((size_t)G4_ * I_ * 2);
    _Float16* whhf = (_Float16*)alloc((size_t)G4_ * H_ * 2);
    _Float16* wfcf = (_Float16*)alloc((size_t)O_ * H_ * 2);
    float*    bias = (float*)   alloc((size_t)G4_ * 4);
    _Float16* h0   = (_Float16*)alloc((size_t)B_ * H_ * 2);
    _Float16* h1   = (_Float16*)alloc((size_t)B_ * H_ * 2);
    int*      sync = (int*)     alloc(256);
    _Float16* xgf  = (_Float16*)alloc((size_t)S_ * B_ * G4_ * 2);   // 256 MB, optional
    const bool use_xg = (off <= ws_size);
    (void)n_in; (void)in_sizes; (void)out_size;

    // one-time conversions + bias fold + barrier init
    cvt_f16_kernel<<<16384, 256, 0, stream>>>(x, xf, (long)B_ * S_ * I_);
    cvt_f16_kernel<<<4096, 256, 0, stream>>>(Wih, wihf, (long)G4_ * I_);
    cvt_f16_kernel<<<4096, 256, 0, stream>>>(Whh, whhf, (long)G4_ * H_);
    cvt_f16_kernel<<<512,  256, 0, stream>>>(Wfc, wfcf, (long)O_ * H_);
    bias_sum_kernel<<<(G4_ + 255) / 256, 256, 0, stream>>>(bih, bhh, bias, G4_);
    init_sync_kernel<<<1, 1, 0, stream>>>(sync);

    dim3 grid(H_ / 64, B_ / 16);
    if (use_xg) {
        // fully-parallel input projection (fills the machine: 32768 blocks)
        xg_precompute_kernel<<<dim3(G4_ / 256, S_ * (B_ / 16)), 128, 0, stream>>>(xf, wihf, xgf);
        // serial path is now only the h-GEMM per step
        lstm_persistent_xg_kernel<<<grid, 128, 0, stream>>>(xgf, whhf, bias,
                                                            h0, h1, sync, sync + 1);
    } else {
        lstm_persistent_kernel<<<grid, 128, 0, stream>>>(xf, wihf, whhf, bias,
                                                         h0, h1, sync, sync + 1);
    }

    // S_ = 512 is even -> final h lives in h0
    fc_kernel<<<B_ / 16, 128, 0, stream>>>(h0, wfcf, bfc, out);
}
